// CNNEncoder_51067161149915
// MI455X (gfx1250) — matrix operand order repaired and checked by reading, compiled, not run on hardware
//
#include <hip/hip_runtime.h>
#include <hip/hip_bf16.h>
#include <cstddef>

typedef __attribute__((ext_vector_type(16))) _Float16 v16h;
typedef __attribute__((ext_vector_type(8)))  _Float16 v8h;
typedef __attribute__((ext_vector_type(4)))  _Float16 v4h;
typedef __attribute__((ext_vector_type(8)))  float    v8f;

#define LSEQ 2048
#define NB   32
#define DIST 10
#define PADW 16
#define LPAD (LSEQ + 2 * PADW)      // 2080 positions per padded sample
#define SLAB 512                    // rows per bn_stats1 slab
#define NSLAB ((NB * LSEQ) / SLAB)  // 128

// ---------------------------------------------------------------------------
// 0) Zero the guard rows of the channel-last padded activation buffer for a
//    given channel count C (layout [B][LPAD][C]).
// ---------------------------------------------------------------------------
__global__ void zero_pads_clast(_Float16* __restrict__ h, int C)
{
    const int q = blockIdx.x * blockDim.x + threadIdx.x;
    const int qreg = C * PADW / 4;            // v4h quads per guard region
    const int qpb  = 2 * qreg;
    if (q >= NB * qpb) return;
    const int b = q / qpb, r = q - b * qpb;
    const int half = r / qreg, rq = r - half * qreg;
    const size_t baseq = ((size_t)b * LPAD * C + (size_t)half * (PADW + LSEQ) * C) / 4;
    ((v4h*)h)[baseq + rq] = (v4h){};
}

// ---------------------------------------------------------------------------
// 1) Peak finding + per-sample standardization -> f16 conv input ([B][LPAD][1])
// ---------------------------------------------------------------------------
__global__ __launch_bounds__(256)
void peaks_norm(const float* __restrict__ x, _Float16* __restrict__ h16)
{
    __shared__ float xs[LSEQ];
    __shared__ unsigned char cand[LSEQ], keep[LSEQ], proc[LSEQ];
    __shared__ float red[256];
    const int b = blockIdx.x, tid = threadIdx.x;
    const float* xr = x + (size_t)b * LSEQ;

    for (int i = tid; i < LSEQ; i += 256) {
        xs[i] = xr[i]; cand[i] = 0; keep[i] = 0; proc[i] = 0;
    }
    __syncthreads();

    float mx = -INFINITY;
    for (int i = tid; i < LSEQ; i += 256) mx = fmaxf(mx, xs[i]);
    red[tid] = mx; __syncthreads();
    for (int st = 128; st > 0; st >>= 1) {
        if (tid < st) red[tid] = fmaxf(red[tid], red[tid + st]);
        __syncthreads();
    }
    mx = red[0]; __syncthreads();

    float sm = 0.f;
    for (int i = tid; i < LSEQ; i += 256) sm += xs[i];
    red[tid] = sm; __syncthreads();
    for (int st = 128; st > 0; st >>= 1) {
        if (tid < st) red[tid] += red[tid + st];
        __syncthreads();
    }
    const float m = red[0] * (1.0f / LSEQ); __syncthreads();

    float qq = 0.f;
    for (int i = tid; i < LSEQ; i += 256) qq += xs[i] * xs[i];
    red[tid] = qq; __syncthreads();
    for (int st = 128; st > 0; st >>= 1) {
        if (tid < st) red[tid] += red[tid + st];
        __syncthreads();
    }
    const float ss = red[0]; __syncthreads();
    const float sdev = sqrtf(fmaxf((ss - (float)LSEQ * m * m) / (float)(LSEQ - 1), 0.0f));

    const float hthr = 0.1f * mx;
    for (int i = tid; i < LSEQ; i += 256)
        if (i >= 1 && i <= LSEQ - 2)
            cand[i] = (xs[i] > xs[i - 1]) && (xs[i] > xs[i + 1]) && (xs[i] >= hthr);
    __syncthreads();

    if (tid == 0) {
        for (;;) {
            float best = -INFINITY; int bi = -1;
            for (int i = 1; i < LSEQ - 1; ++i)
                if (cand[i] && !proc[i] && xs[i] > best) { best = xs[i]; bi = i; }
            if (bi < 0) break;
            proc[bi] = 1;
            bool ok = true;
            const int lo = bi - (DIST - 1) < 0 ? 0 : bi - (DIST - 1);
            const int hi = bi + (DIST - 1) > LSEQ - 1 ? LSEQ - 1 : bi + (DIST - 1);
            for (int j = lo; j <= hi; ++j) if (keep[j]) { ok = false; break; }
            if (ok) keep[bi] = 1;
        }
    }
    __syncthreads();

    const float pthr = 0.05f * mx;
    for (int i = tid; i < LSEQ; i += 256) {
        if (!keep[i]) continue;
        const float xi = xs[i];
        float lmin = xi;
        for (int j = i - 1; j >= 0 && xs[j] <= xi; --j) lmin = fminf(lmin, xs[j]);
        float rmin = xi;
        for (int j = i + 1; j < LSEQ && xs[j] <= xi; ++j) rmin = fminf(rmin, xs[j]);
        if (xi - fmaxf(lmin, rmin) < pthr) keep[i] = 0;
    }
    __syncthreads();

    const float inv = 1.0f / (sdev + 1e-5f);
    for (int i = tid; i < LSEQ; i += 256) {
        const float w = keep[i] ? 1.1f : 1.0f;
        h16[(size_t)b * LPAD + PADW + i] = (_Float16)(w * (xs[i] - m) * inv);
    }
}

// ---------------------------------------------------------------------------
// 2) Implicit-GEMM conv1d via v_wmma_f32_16x16x32_f16, channel-last.
//    K' = kk*CIN + ci  =>  B address is LINEAR in K' (kk*CIN+ci0 == base),
//    so both A and B streams are one base pointer bumped 32 elems per chunk
//    with compile-time immediate offsets. MT-way M register blocking.
// ---------------------------------------------------------------------------
template<int KSIZE, int CIN, int MT>
__global__ __launch_bounds__(32)
void conv1d_wmma(const _Float16* __restrict__ w16,   // [Cout][KPAD] K'-ordered
                 const float* __restrict__ bias,     // [Cout]
                 const _Float16* __restrict__ in16,  // [B][LPAD][CIN], zero guards
                 float* __restrict__ y,              // [B][LSEQ][Cout]
                 int Cout)
{
    constexpr int KTOT = CIN * KSIZE;
    constexpr int PAD  = KSIZE / 2;
    constexpr int NCH  = (KTOT + 31) >> 5;
    constexpr int KPAD = NCH << 5;

    const int lane  = threadIdx.x;          // 0..31 (wave32)
    const int halfh = lane >> 4;            // 0: lanes 0-15, 1: lanes 16-31
    const int nCol  = (blockIdx.x << 4) | (lane & 15);  // global position
    const int b     = nCol >> 11;
    const int l     = nCol & (LSEQ - 1);
    const int coBase = blockIdx.y * (MT * 16);
    const int mRow   = lane & 15;

    const _Float16* __restrict__ inP =
        in16 + ((size_t)b * LPAD + l + (PADW - PAD)) * CIN;

    // Linear base pointers; all other addressing via constant offsets.
    const _Float16* __restrict__ bp = inP + (halfh << 4);
    const _Float16* __restrict__ ap =
        w16 + (size_t)(coBase + mRow) * KPAD + (halfh << 4);

    v8f acc[MT] = {};
    for (int ch = 0; ch < NCH; ++ch) {
        v16h bf;
        if constexpr (CIN >= 16) {
            const v8h b0 = *(const v8h*)(bp);
            const v8h b1 = *(const v8h*)(bp + 8);
            bf = __builtin_shufflevector(b0, b1,
                0, 1, 2, 3, 4, 5, 6, 7, 8, 9, 10, 11, 12, 13, 14, 15);
        } else {  // CIN == 1 (conv0, single chunk): scalar clamped path
            const int base = (ch << 5) + (halfh << 4);
#pragma unroll
            for (int e = 0; e < 16; ++e) {
                int K = base + e;
                K = (K < KTOT) ? K : (KTOT - 1);  // matching A entries are zero
                bf[e] = inP[K];
            }
        }
#pragma unroll
        for (int mt = 0; mt < MT; ++mt) {
            const v8h alo = *(const v8h*)(ap + (size_t)mt * 16 * KPAD);
            const v8h ahi = *(const v8h*)(ap + (size_t)mt * 16 * KPAD + 8);
            const v16h a = __builtin_shufflevector(alo, ahi,
                0, 1, 2, 3, 4, 5, 6, 7, 8, 9, 10, 11, 12, 13, 14, 15);
            acc[mt] = __builtin_amdgcn_wmma_f32_16x16x32_f16(
                false, a, false, bf, (short)0, acc[mt], false, false);
        }
        bp += 32;
        ap += 32;
    }
    // D: lane has fixed position l; rows r -> co = co0 + r contiguous in memory
    float* __restrict__ outRow = y + ((size_t)b * LSEQ + l) * Cout;
#pragma unroll
    for (int mt = 0; mt < MT; ++mt) {
        const int co0 = coBase + (mt << 4) + (halfh << 3);
        const float4 ba = *(const float4*)(bias + co0);
        const float4 bb4 = *(const float4*)(bias + co0 + 4);
        float4 o0, o1;
        o0.x = acc[mt][0] + ba.x;  o0.y = acc[mt][1] + ba.y;
        o0.z = acc[mt][2] + ba.z;  o0.w = acc[mt][3] + ba.w;
        o1.x = acc[mt][4] + bb4.x; o1.y = acc[mt][5] + bb4.y;
        o1.z = acc[mt][6] + bb4.z; o1.w = acc[mt][7] + bb4.w;
        *(float4*)(outRow + co0)     = o0;
        *(float4*)(outRow + co0 + 4) = o1;
    }
}

// ---------------------------------------------------------------------------
// 3) Batchnorm over channel-last y: two-stage deterministic reduction.
// ---------------------------------------------------------------------------
__global__ __launch_bounds__(256)
void bn_stats1(const float* __restrict__ y, float* __restrict__ part, int C)
{
    __shared__ float s1[256], s2[256];
    const int G = 256 / C;                  // C in {64,128,256}
    const int c = threadIdx.x % C;
    const int g = threadIdx.x / C;
    const size_t r0 = (size_t)blockIdx.x * SLAB;
    float sum = 0.f, sq = 0.f;
    for (int r = g; r < SLAB; r += G) {
        const float v = y[(r0 + r) * C + c];
        sum += v; sq += v * v;
    }
    s1[threadIdx.x] = sum; s2[threadIdx.x] = sq;
    for (int hg = G >> 1; hg > 0; hg >>= 1) {
        __syncthreads();
        if (threadIdx.x < hg * C) {
            s1[threadIdx.x] += s1[threadIdx.x + hg * C];
            s2[threadIdx.x] += s2[threadIdx.x + hg * C];
        }
    }
    __syncthreads();
    if (threadIdx.x < C) {
        part[(size_t)(blockIdx.x * 2) * C + threadIdx.x]     = s1[threadIdx.x];
        part[(size_t)(blockIdx.x * 2 + 1) * C + threadIdx.x] = s2[threadIdx.x];
    }
}

__global__ __launch_bounds__(128)
void bn_stats2(const float* __restrict__ part, const float* __restrict__ g,
               const float* __restrict__ beta, float* __restrict__ scale,
               float* __restrict__ shift, int C)
{
    __shared__ float s1[128], s2[128];
    const int c = blockIdx.x;
    float sum = 0.f, sq = 0.f;
    for (int s = threadIdx.x; s < NSLAB; s += 128) {
        sum += part[(size_t)(s * 2) * C + c];
        sq  += part[(size_t)(s * 2 + 1) * C + c];
    }
    s1[threadIdx.x] = sum; s2[threadIdx.x] = sq;
    __syncthreads();
    for (int st = 64; st > 0; st >>= 1) {
        if (threadIdx.x < st) {
            s1[threadIdx.x] += s1[threadIdx.x + st];
            s2[threadIdx.x] += s2[threadIdx.x + st];
        }
        __syncthreads();
    }
    if (threadIdx.x == 0) {
        const float n = (float)(NB * LSEQ);
        const float mean = s1[0] / n;
        const float var = s2[0] / n - mean * mean;
        const float inv = rsqrtf(var + 1e-5f);
        const float sc = g[c] * inv;
        scale[c] = sc;
        shift[c] = beta[c] - mean * sc;
    }
}

__global__ void bn_relu_clast(const float* __restrict__ y, const float* __restrict__ scale,
                              const float* __restrict__ shift, _Float16* __restrict__ h,
                              int C, int cShift, int totalq)
{
    const int i = blockIdx.x * blockDim.x + threadIdx.x;   // quad index
    if (i >= totalq) return;
    const int e0 = i << 2;
    const int row = e0 >> cShift;            // b*LSEQ + l
    const int c0 = e0 & (C - 1);             // multiple of 4
    const int b = row >> 11, l = row & (LSEQ - 1);
    const float4 v = ((const float4*)y)[i];
    const float4 sc = *(const float4*)(scale + c0);
    const float4 sh = *(const float4*)(shift + c0);
    v4h o;
    o[0] = (_Float16)fmaxf(v.x * sc.x + sh.x, 0.0f);
    o[1] = (_Float16)fmaxf(v.y * sc.y + sh.y, 0.0f);
    o[2] = (_Float16)fmaxf(v.z * sc.z + sh.z, 0.0f);
    o[3] = (_Float16)fmaxf(v.w * sc.w + sh.w, 0.0f);
    ((v4h*)h)[((((size_t)b * LPAD + PADW + l) << cShift) + c0) >> 2] = o;
}

// ---------------------------------------------------------------------------
// 4) Mean pool over L (channel-last, C=256), then the two 64x256 heads
// ---------------------------------------------------------------------------
__global__ __launch_bounds__(256)
void pool_mean_clast(const _Float16* __restrict__ h, float* __restrict__ feat)
{
    const int b = blockIdx.x, c = threadIdx.x;      // C == 256
    const _Float16* __restrict__ p = h + ((size_t)b * LPAD + PADW) * 256 + c;
    float sum = 0.f;
    for (int l = 0; l < LSEQ; ++l) sum += (float)p[(size_t)l * 256];
    feat[b * 256 + c] = sum * (1.0f / LSEQ);
}

__global__ void heads(const float* __restrict__ feat,
                      const float* __restrict__ wm, const float* __restrict__ bm,
                      const float* __restrict__ wv, const float* __restrict__ bv,
                      float* __restrict__ out)
{
    const int i = blockIdx.x * blockDim.x + threadIdx.x;
    if (i >= 2 * NB * 64) return;
    const int which = i >> 11;
    const int r = i & 2047;
    const int b = r >> 6, j = r & 63;
    const float* __restrict__ w  = which ? wv : wm;
    const float* __restrict__ bi = which ? bv : bm;
    const float4* __restrict__ f  = (const float4*)(feat + b * 256);
    const float4* __restrict__ wr = (const float4*)(w + j * 256);
    float acc = bi[j];
    for (int k = 0; k < 64; ++k) {
        const float4 a = f[k], c = wr[k];
        acc += a.x * c.x + a.y * c.y + a.z * c.z + a.w * c.w;
    }
    out[i] = acc;
}

// pack conv weights [Cout][Cin][Ks] f32 -> [Cout][Kpad] f16, K' = kk*CIN + ci
__global__ void pack_w16(const float* __restrict__ s, _Float16* __restrict__ d,
                         int Cin, int Ks, int Kpad, int n)
{
    const int i = blockIdx.x * blockDim.x + threadIdx.x;
    if (i >= n) return;
    const int co = i / Kpad, Kp = i - co * Kpad;
    const int Ktot = Cin * Ks;
    _Float16 v = (_Float16)0.0f;
    if (Kp < Ktot) {
        const int kk = Kp / Cin, ci = Kp - kk * Cin;
        v = (_Float16)s[(co * Cin + ci) * Ks + kk];
    }
    d[i] = v;
}

// ---------------------------------------------------------------------------
// Host launcher
// ---------------------------------------------------------------------------
extern "C" void kernel_launch(void* const* d_in, const int* in_sizes, int n_in,
                              void* d_out, int out_size, void* d_ws, size_t ws_size,
                              hipStream_t stream)
{
    (void)in_sizes; (void)n_in; (void)out_size; (void)ws_size;

    const float* x = (const float*)d_in[0];
    const int CIN[6]  = {1, 64, 64, 128, 128, 256};
    const int COUT[6] = {64, 64, 128, 128, 256, 256};
    const int KS[6]   = {5, 5, 15, 15, 25, 25};
    const int CSH[6]  = {6, 6, 7, 7, 8, 8};   // log2(COUT)
    const int MT[6]   = {4, 4, 8, 8, 8, 8};   // M-tiles per wave

    // workspace carve-up
    char* ws = (char*)d_ws;
    _Float16* h16 = (_Float16*)ws;                       // [B][LPAD][<=256] f16
    size_t off = (size_t)NB * LPAD * 256 * sizeof(_Float16);
    off = (off + 255) & ~(size_t)255;
    float* y32 = (float*)(ws + off);                     // [B][LSEQ][<=256] f32
    off += (size_t)NB * LSEQ * 256 * sizeof(float);
    _Float16* w16 = (_Float16*)(ws + off);               // K'-ordered f16 weights
    size_t wOff[6], wAcc = 0;
    int KPAD[6];
    for (int i = 0; i < 6; ++i) {
        KPAD[i] = ((CIN[i] * KS[i] + 31) / 32) * 32;
        wOff[i] = wAcc;
        wAcc += (size_t)COUT[i] * KPAD[i];
    }
    off += wAcc * sizeof(_Float16);
    off = (off + 255) & ~(size_t)255;
    float* part  = (float*)(ws + off); off += (size_t)NSLAB * 2 * 256 * sizeof(float);
    float* scale = (float*)(ws + off); off += 256 * sizeof(float);
    float* shift = (float*)(ws + off); off += 256 * sizeof(float);
    float* feat  = (float*)(ws + off); off += (size_t)NB * 256 * sizeof(float);

    // pack conv weights to zero-padded, K'-ordered f16 rows
    for (int i = 0; i < 6; ++i) {
        const int n = COUT[i] * KPAD[i];
        pack_w16<<<(n + 255) / 256, 256, 0, stream>>>(
            (const float*)d_in[1 + 4 * i], w16 + wOff[i], CIN[i], KS[i], KPAD[i], n);
    }

    // guard rows for conv0's input layout (C=1), then peaks + standardize
    {
        const int n = NB * 2 * 1 * PADW / 4;
        zero_pads_clast<<<(n + 255) / 256, 256, 0, stream>>>(h16, 1);
    }
    peaks_norm<<<NB, 256, 0, stream>>>(x, h16);

    for (int i = 0; i < 6; ++i) {
        const float* cb = (const float*)d_in[2 + 4 * i];
        const float* bg = (const float*)d_in[3 + 4 * i];
        const float* bb = (const float*)d_in[4 + 4 * i];
        const dim3 grid((NB * LSEQ) / 16, COUT[i] / (16 * MT[i]));
        switch (i) {
        case 0: conv1d_wmma<5, 1, 4>    <<<grid, 32, 0, stream>>>(w16 + wOff[0], cb, h16, y32, COUT[0]); break;
        case 1: conv1d_wmma<5, 64, 4>   <<<grid, 32, 0, stream>>>(w16 + wOff[1], cb, h16, y32, COUT[1]); break;
        case 2: conv1d_wmma<15, 64, 8>  <<<grid, 32, 0, stream>>>(w16 + wOff[2], cb, h16, y32, COUT[2]); break;
        case 3: conv1d_wmma<15, 128, 8> <<<grid, 32, 0, stream>>>(w16 + wOff[3], cb, h16, y32, COUT[3]); break;
        case 4: conv1d_wmma<25, 128, 8> <<<grid, 32, 0, stream>>>(w16 + wOff[4], cb, h16, y32, COUT[4]); break;
        case 5: conv1d_wmma<25, 256, 8> <<<grid, 32, 0, stream>>>(w16 + wOff[5], cb, h16, y32, COUT[5]); break;
        }
        bn_stats1<<<NSLAB, 256, 0, stream>>>(y32, part, COUT[i]);
        bn_stats2<<<COUT[i], 128, 0, stream>>>(part, bg, bb, scale, shift, COUT[i]);
        const int totalq = NB * LSEQ * COUT[i] / 4;
        bn_relu_clast<<<(totalq + 255) / 256, 256, 0, stream>>>(
            y32, scale, shift, h16, COUT[i], CSH[i], totalq);
        if (i < 5) {   // guards for the next conv's input layout (C = COUT[i])
            const int n = NB * 2 * COUT[i] * PADW / 4;
            zero_pads_clast<<<(n + 255) / 256, 256, 0, stream>>>(h16, COUT[i]);
        }
    }

    pool_mean_clast<<<NB, 256, 0, stream>>>(h16, feat);
    heads<<<(2 * NB * 64 + 255) / 256, 256, 0, stream>>>(
        feat, (const float*)d_in[25], (const float*)d_in[26],
        (const float*)d_in[27], (const float*)d_in[28], (float*)d_out);
}
